// GNNOnlyModel_27685359190658
// MI455X (gfx1250) — compile-verified, matrix operand order checked by reading, not run on hardware
//
#include <hip/hip_runtime.h>
#include <hip/hip_bf16.h>
#include <math.h>

typedef _Float16 f16;
typedef __attribute__((ext_vector_type(16))) _Float16 v16h;
typedef __attribute__((ext_vector_type(8)))  _Float16 v8h;
typedef __attribute__((ext_vector_type(8)))  float    v8f;

#define NN   100000
#define NE   400000
#define NG   4096
#define HID  256
#define HEADS 8
#define HC   32

// ---------- float <-> monotone int key (for atomic max on signed floats) ----
__device__ __forceinline__ int fkey(float f) {
  int b = __float_as_int(f);
  return b < 0 ? (b ^ 0x7FFFFFFF) : b;
}
__device__ __forceinline__ float funkey(int k) {
  int b = k < 0 ? (k ^ 0x7FFFFFFF) : k;
  return __int_as_float(b);
}

// ---------------------------- fills ----------------------------------------
__global__ void k_fill_f32(float* p, long n, float v) {
  long i = (long)blockIdx.x * blockDim.x + threadIdx.x;
  if (i < n) p[i] = v;
}
__global__ void k_fill_i32(int* p, long n, int v) {
  long i = (long)blockIdx.x * blockDim.x + threadIdx.x;
  if (i < n) p[i] = v;
}

// ---------------- weight convert + transpose: W[KxN=256] -> WT[256xK] f16 ---
__global__ void k_convert_T(const float* __restrict__ W, f16* __restrict__ WT, int K) {
  int i = blockIdx.x * blockDim.x + threadIdx.x;
  if (i >= 256 * K) return;
  int n = i / K, k = i % K;
  WT[i] = (f16)W[(long)k * 256 + n];
}

// ---------------- node embedding: h = concat(emb[idx], x[:,1:]@W + b) -------
__global__ void k_embed(const float* __restrict__ x, const float* __restrict__ emb,
                        const float* __restrict__ W, const float* __restrict__ b,
                        f16* __restrict__ h16) {
  int i = blockIdx.x * blockDim.x + threadIdx.x;
  if (i >= NN * HID) return;
  int node = i >> 8, c = i & 255;
  float v;
  if (c < 128) {
    int idx = (int)x[(long)node * 9];
    v = emb[(long)idx * 128 + c];
  } else {
    int co = c - 128;
    float s = b[co];
    const float* xr = x + (long)node * 9 + 1;
#pragma unroll
    for (int j = 0; j < 8; ++j) s += xr[j] * W[j * 128 + co];
    v = s;
  }
  h16[i] = (f16)v;
}

// --------- WMMA GEMM: C[M,256] = A[M,K](f16) * B[K,256], BT = f16[256,K] ----
// Block = 8 waves covering a 64-row x 256-col panel: 4 M-tiles x 2 N-halves.
// Each wave owns a 16x128 panel -> only 8 live v8f accumulators (64 VGPRs),
// avoiding scratch spills. B is staged through LDS in 32KB K-slices shared by
// all waves; each B element is read from LDS once per consuming wave.
template <int K>
__global__ void __launch_bounds__(256, 2)
gemm_wmma(const f16* __restrict__ A, const f16* __restrict__ BT,
          float* __restrict__ C, int Mtiles) {
  __shared__ f16 Bs[256 * 64];            // [n=256][k-slice=64] : 32 KB
  int wave = threadIdx.x >> 5;
  int lane = threadIdx.x & 31;
  int mtile = blockIdx.x * 4 + (wave >> 1);
  if (mtile >= Mtiles) mtile = Mtiles - 1;   // clamp: keeps barriers uniform
  int nbase = (wave & 1) * 8;                // this wave's 8 N-tiles
  int lo = lane & 15;
  int hi = (lane >> 4) & 1;
  int t = threadIdx.x;

  v8f acc[8];
#pragma unroll
  for (int nt = 0; nt < 8; ++nt) acc[nt] = (v8f){};

  const f16* arow = A + (long)(mtile * 16 + lo) * K;

  for (int ks = 0; ks < K; ks += 64) {
    // ---- stage B slice [256 n][64 k] into LDS (each thread: one n-row) ----
    __syncthreads();                       // previous slice fully consumed
    {
      const f16* gsrc = BT + (long)t * K + ks;
      f16* ldst = Bs + t * 64;
#pragma unroll
      for (int j = 0; j < 8; ++j)
        *(v8h*)(ldst + j * 8) = *(const v8h*)(gsrc + j * 8);
      if (ks + 64 < K)                     // prefetch next slice into caches
        __builtin_prefetch(BT + (long)t * K + ks + 64, 0, 1);
    }
    __syncthreads();

    // ---- A fragments for the two 32-K tiles of this slice -----------------
    v16h a0, a1;
    {
      v8h l0 = *(const v8h*)(arow + ks + hi * 8);
      v8h l1 = *(const v8h*)(arow + ks + hi * 8 + 16);
      v8h m0 = *(const v8h*)(arow + ks + 32 + hi * 8);
      v8h m1 = *(const v8h*)(arow + ks + 32 + hi * 8 + 16);
#pragma unroll
      for (int j = 0; j < 8; ++j) {
        a0[j] = l0[j]; a0[j + 8] = l1[j];
        a1[j] = m0[j]; a1[j + 8] = m1[j];
      }
    }

    // ---- 8 N-tiles x 2 K-tiles of WMMA fed from LDS -----------------------
#pragma unroll
    for (int nt = 0; nt < 8; ++nt) {
      const f16* brow = Bs + ((nbase + nt) * 16 + lo) * 64;
      v8h b0 = *(const v8h*)(brow + hi * 8);
      v8h b1 = *(const v8h*)(brow + hi * 8 + 16);
      v8h b2 = *(const v8h*)(brow + 32 + hi * 8);
      v8h b3 = *(const v8h*)(brow + 32 + hi * 8 + 16);
      v16h bf0, bf1;
#pragma unroll
      for (int j = 0; j < 8; ++j) {
        bf0[j] = b0[j]; bf0[j + 8] = b1[j];
        bf1[j] = b2[j]; bf1[j + 8] = b3[j];
      }
      acc[nt] = __builtin_amdgcn_wmma_f32_16x16x32_f16(false, a0, false, bf0,
                                                       (short)0, acc[nt], false, false);
      acc[nt] = __builtin_amdgcn_wmma_f32_16x16x32_f16(false, a1, false, bf1,
                                                       (short)0, acc[nt], false, false);
    }
  }

  // ---- store: VGPR r -> row mtile*16 + r + hi*8, col (nbase+nt)*16 + lo ---
#pragma unroll
  for (int nt = 0; nt < 8; ++nt) {
    float* crow = C + (long)(mtile * 16 + hi * 8) * 256 + (nbase + nt) * 16 + lo;
#pragma unroll
    for (int r = 0; r < 8; ++r) crow[(long)r * 256] = acc[nt][r];
  }
}

// ------- edge attention LUT: a_e[5][8] (EF=5 distinct edge features) --------
__global__ void k_edge_table(const float* __restrict__ We, const float* __restrict__ att_e,
                             const float* __restrict__ edge_emb, float* __restrict__ aetab) {
  __shared__ float v[256 * 8];
  int j = threadIdx.x;
  for (int hd = 0; hd < 8; ++hd) {
    float s = 0.f;
    const float* wr = We + (long)j * 256 + hd * 32;
    const float* ar = att_e + hd * 32;
#pragma unroll
    for (int c = 0; c < 32; ++c) s += wr[c] * ar[c];
    v[j * 8 + hd] = s;
  }
  __syncthreads();
  if (j < 40) {
    int i = j >> 3, hd = j & 7;
    float s = 0.f;
    const float* er = edge_emb + (long)i * 256;
    for (int k = 0; k < 256; ++k) s += er[k] * v[k * 8 + hd];
    aetab[i * 8 + hd] = s;
  }
}

// ---------------- per-node attention scores --------------------------------
__global__ void k_att_scores(const float* __restrict__ xh, const float* __restrict__ att_s,
                             const float* __restrict__ att_d, float* __restrict__ asrc,
                             float* __restrict__ adst) {
  int i = blockIdx.x * blockDim.x + threadIdx.x;
  if (i >= NN * HEADS) return;
  int node = i >> 3, hd = i & 7;
  const float* xr = xh + (long)node * 256 + hd * 32;
  float s = 0.f, d = 0.f;
#pragma unroll
  for (int c = 0; c < 32; ++c) {
    float v = xr[c];
    s += v * att_s[hd * 32 + c];
    d += v * att_d[hd * 32 + c];
  }
  asrc[i] = s;
  adst[i] = d;
}

// ---------------- edge alpha + segment max ---------------------------------
__global__ void k_alpha_max(const float* __restrict__ asrc, const float* __restrict__ adst,
                            const float* __restrict__ aetab, const int* __restrict__ src,
                            const int* __restrict__ dst, const int* __restrict__ attr,
                            float* __restrict__ alpha, int* __restrict__ amax) {
  int i = blockIdx.x * blockDim.x + threadIdx.x;
  if (i >= NE * HEADS) return;
  int e = i >> 3, hd = i & 7;
  int s = src[e], d = dst[e], a = attr[e];
  float al = asrc[s * 8 + hd] + adst[d * 8 + hd] + aetab[a * 8 + hd];
  al = al > 0.f ? al : 0.2f * al;        // leaky relu
  alpha[i] = al;
  atomicMax(&amax[d * 8 + hd], fkey(al));
}

// ---------------- exp(alpha - max) + segment sum ---------------------------
__global__ void k_exp_sum(float* __restrict__ alpha, const int* __restrict__ amax,
                          const int* __restrict__ dst, float* __restrict__ denom) {
  int i = blockIdx.x * blockDim.x + threadIdx.x;
  if (i >= NE * HEADS) return;
  int e = i >> 3, hd = i & 7;
  float am = funkey(amax[dst[e] * 8 + hd]);
  float ex = expf(alpha[i] - am);
  alpha[i] = ex;                         // reuse buffer
  atomicAdd(&denom[dst[e] * 8 + hd], ex);
}

// ---------------- weighted aggregation to destination nodes ----------------
__global__ void k_aggregate(const float* __restrict__ ex, const float* __restrict__ denom,
                            const int* __restrict__ src, const int* __restrict__ dst,
                            const float* __restrict__ xh, float* __restrict__ agg) {
  int i = blockIdx.x * blockDim.x + threadIdx.x;
  if (i >= NE * HEADS) return;
  int e = i >> 3, hd = i & 7;
  int d = dst[e], s = src[e];
  float w = ex[i] / (denom[d * 8 + hd] + 1e-16f);
  const float* xr = xh + (long)s * 256 + hd * 32;
  float* orow = agg + (long)d * 256 + hd * 32;
#pragma unroll
  for (int c = 0; c < 32; ++c) atomicAdd(&orow[c], xr[c] * w);
}

// ---------------- BatchNorm (biased var) + ReLU ----------------------------
__global__ void k_bn_stats(const float* __restrict__ X, int rows,
                           float* __restrict__ sums, float* __restrict__ sumsq) {
  int ch = threadIdx.x;
  int r0 = blockIdx.x * 128;
  float s = 0.f, s2 = 0.f;
  for (int r = 0; r < 128; ++r) {
    int row = r0 + r;
    if (row < rows) {
      float v = X[(long)row * 256 + ch];
      s += v; s2 += v * v;
    }
  }
  atomicAdd(&sums[ch], s);
  atomicAdd(&sumsq[ch], s2);
}
__global__ void k_bn_final(const float* sums, const float* sumsq, float n,
                           const float* g, const float* b,
                           float* scale, float* shift) {
  int ch = threadIdx.x;
  float m = sums[ch] / n;
  float var = sumsq[ch] / n - m * m;
  float a = g[ch] * rsqrtf(var + 1e-5f);
  scale[ch] = a;
  shift[ch] = b[ch] - m * a;
}
__global__ void k_bn_apply(const float* __restrict__ X, const float* __restrict__ scale,
                           const float* __restrict__ shift, long total,
                           float* __restrict__ Yf, f16* __restrict__ Yh) {
  long i = (long)blockIdx.x * blockDim.x + threadIdx.x;
  if (i >= total) return;
  int ch = (int)(i & 255);
  float y = X[i] * scale[ch] + shift[ch];
  y = y > 0.f ? y : 0.f;
  Yf[i] = y;
  Yh[i] = (f16)y;
}

// ---------------- graph pooling --------------------------------------------
__global__ void k_pool_cnt(const int* __restrict__ batch, float* __restrict__ cnt) {
  int i = blockIdx.x * blockDim.x + threadIdx.x;
  if (i < NN) atomicAdd(&cnt[batch[i]], 1.0f);
}
__global__ void k_pool_scatter(const float* __restrict__ h, const int* __restrict__ batch,
                               float* __restrict__ psum, unsigned int* __restrict__ pmax) {
  long i = (long)blockIdx.x * blockDim.x + threadIdx.x;
  if (i >= (long)NN * 256) return;
  int node = (int)(i >> 8), ch = (int)(i & 255);
  int g = batch[node];
  float v = h[i];
  atomicAdd(&psum[(long)g * 256 + ch], v);
  atomicMax(&pmax[(long)g * 256 + ch], __float_as_uint(v)); // v >= 0 post-relu
}
__global__ void k_pool_gather(const float* __restrict__ psum, const unsigned int* __restrict__ pmax,
                              const float* __restrict__ cnt, f16* __restrict__ g16) {
  int i = blockIdx.x * blockDim.x + threadIdx.x;
  if (i >= NG * 256) return;
  int g = i >> 8, ch = i & 255;
  float mean = psum[i] / fmaxf(cnt[g], 1.0f);
  float mx = __uint_as_float(pmax[i]);   // init 0 handles empty graphs
  g16[(long)g * 512 + ch]       = (f16)mean;
  g16[(long)g * 512 + 256 + ch] = (f16)mx;
}

// ---------------- regressor helpers ----------------------------------------
__global__ void k_axpy_row(float* __restrict__ r, const float* __restrict__ Nv,
                           const float* __restrict__ Wrow) {
  int i = blockIdx.x * blockDim.x + threadIdx.x;
  if (i >= NG * 256) return;
  r[i] += Nv[i >> 8] * Wrow[i & 255];
}
__global__ void k_reg_out(const float* __restrict__ r, const float* __restrict__ Wo,
                          const float* __restrict__ bo, float* __restrict__ out) {
  int g = blockIdx.x * blockDim.x + threadIdx.x;
  if (g >= NG) return;
  float s = bo[0];
  const float* rr = r + (long)g * 256;
  for (int c = 0; c < 256; ++c) s += rr[c] * Wo[c];
  out[g] = s;
}

// ============================================================================
extern "C" void kernel_launch(void* const* d_in, const int* in_sizes, int n_in,
                              void* d_out, int out_size, void* d_ws, size_t ws_size,
                              hipStream_t stream) {
  (void)in_sizes; (void)n_in; (void)out_size; (void)ws_size;
  const float* x        = (const float*)d_in[0];
  const int*   src      = (const int*)d_in[1];        // edge_index[0]
  const int*   dst      = (const int*)d_in[1] + NE;   // edge_index[1]
  const int*   eattr    = (const int*)d_in[2];
  const int*   batch    = (const int*)d_in[3];
  const float* Nvals    = (const float*)d_in[4];
  const float* node_emb = (const float*)d_in[5];
  const float* lin_W    = (const float*)d_in[6];
  const float* lin_b    = (const float*)d_in[7];
  const float* edge_emb = (const float*)d_in[8];
  // convs[l]: base 9 + 8*l -> W, We, att_src, att_dst, att_edge, bias, bn_g, bn_b
  const float* reg_in_W    = (const float*)d_in[33];
  const float* reg_in_g    = (const float*)d_in[35];
  const float* reg_in_beta = (const float*)d_in[36];
  // reg_hid[i]: base 37 + 4*i -> W, b, g, beta
  const float* reg_out_W = (const float*)d_in[49];
  const float* reg_out_b = (const float*)d_in[50];
  float* out = (float*)d_out;

  // ---- workspace carve ----
  char* w = (char*)d_ws;
  auto carve = [&](size_t bytes) -> char* {
    char* p = w; w += (bytes + 255) & ~(size_t)255; return p;
  };
  f16*   h16   = (f16*)  carve((size_t)NN * 256 * 2);
  float* xh    = (float*)carve((size_t)NN * 256 * 4);   // also holds post-BN h
  float* agg   = (float*)carve((size_t)NN * 256 * 4);
  float* asrc  = (float*)carve((size_t)NN * 8 * 4);
  float* adst  = (float*)carve((size_t)NN * 8 * 4);
  float* alpha = (float*)carve((size_t)NE * 8 * 4);
  int*   amax  = (int*)  carve((size_t)NN * 8 * 4);
  float* denom = (float*)carve((size_t)NN * 8 * 4);
  float* aetab = (float*)carve(64 * 4);
  float* sums  = (float*)carve(256 * 4);
  float* sumsq = (float*)carve(256 * 4);
  float* scale = (float*)carve(256 * 4);
  float* shift = (float*)carve(256 * 4);
  float* psum  = (float*)carve((size_t)NG * 256 * 4);
  unsigned int* pmax = (unsigned int*)carve((size_t)NG * 256 * 4);
  float* cnt   = (float*)carve((size_t)NG * 4);
  f16*   g16   = (f16*)  carve((size_t)NG * 512 * 2);
  float* rA    = (float*)carve((size_t)NG * 256 * 4);
  f16*   r16   = (f16*)  carve((size_t)NG * 256 * 2);
  f16*   convT0 = (f16*) carve((size_t)3 * 65536 * 2);
  f16*   regInT = (f16*) carve((size_t)256 * 512 * 2);
  f16*   regHT0 = (f16*) carve((size_t)3 * 65536 * 2);

  const int B = 256;
  auto g1 = [&](long n) { return dim3((unsigned)((n + B - 1) / B)); };

  // ---- convert all GEMM weights to transposed f16 once ----
  for (int l = 0; l < 3; ++l)
    k_convert_T<<<g1(65536), B, 0, stream>>>((const float*)d_in[9 + 8 * l], convT0 + (size_t)l * 65536, 256);
  k_convert_T<<<g1(256 * 512), B, 0, stream>>>(reg_in_W, regInT, 512);
  for (int i = 0; i < 3; ++i)
    k_convert_T<<<g1(65536), B, 0, stream>>>((const float*)d_in[37 + 4 * i], regHT0 + (size_t)i * 65536, 256);

  // ---- node embedding ----
  k_embed<<<g1((long)NN * 256), B, 0, stream>>>(x, node_emb, lin_W, lin_b, h16);

  // ---- 3 GAT layers ----
  const int Mtiles = NN / 16;   // 6250
  for (int l = 0; l < 3; ++l) {
    const float* We    = (const float*)d_in[9 + 8 * l + 1];
    const float* att_s = (const float*)d_in[9 + 8 * l + 2];
    const float* att_d = (const float*)d_in[9 + 8 * l + 3];
    const float* att_e = (const float*)d_in[9 + 8 * l + 4];
    const float* bn_g  = (const float*)d_in[9 + 8 * l + 6];
    const float* bn_b  = (const float*)d_in[9 + 8 * l + 7];

    gemm_wmma<256><<<dim3((Mtiles + 3) / 4), B, 0, stream>>>(h16, convT0 + (size_t)l * 65536, xh, Mtiles);
    k_edge_table<<<1, 256, 0, stream>>>(We, att_e, edge_emb, aetab);
    k_att_scores<<<g1((long)NN * 8), B, 0, stream>>>(xh, att_s, att_d, asrc, adst);

    k_fill_i32<<<g1((long)NN * 8), B, 0, stream>>>(amax, (long)NN * 8, 0x80000000);
    k_fill_f32<<<g1((long)NN * 8), B, 0, stream>>>(denom, (long)NN * 8, 0.f);
    k_fill_f32<<<g1((long)NN * 256), B, 0, stream>>>(agg, (long)NN * 256, 0.f);

    k_alpha_max<<<g1((long)NE * 8), B, 0, stream>>>(asrc, adst, aetab, src, dst, eattr, alpha, amax);
    k_exp_sum<<<g1((long)NE * 8), B, 0, stream>>>(alpha, amax, dst, denom);
    k_aggregate<<<g1((long)NE * 8), B, 0, stream>>>(alpha, denom, src, dst, xh, agg);

    // GAT bias cancels inside BatchNorm -> skipped.
    k_fill_f32<<<1, 256, 0, stream>>>(sums, 256, 0.f);
    k_fill_f32<<<1, 256, 0, stream>>>(sumsq, 256, 0.f);
    k_bn_stats<<<dim3((NN + 127) / 128), 256, 0, stream>>>(agg, NN, sums, sumsq);
    k_bn_final<<<1, 256, 0, stream>>>(sums, sumsq, (float)NN, bn_g, bn_b, scale, shift);
    k_bn_apply<<<g1((long)NN * 256), B, 0, stream>>>(agg, scale, shift, (long)NN * 256, xh, h16);
  }

  // ---- graph pooling (mean + max) ----
  k_fill_f32<<<g1((long)NG * 256), B, 0, stream>>>(psum, (long)NG * 256, 0.f);
  k_fill_i32<<<g1((long)NG * 256), B, 0, stream>>>((int*)pmax, (long)NG * 256, 0);
  k_fill_f32<<<g1(NG), B, 0, stream>>>(cnt, NG, 0.f);
  k_pool_cnt<<<g1(NN), B, 0, stream>>>(batch, cnt);
  k_pool_scatter<<<g1((long)NN * 256), B, 0, stream>>>(xh, batch, psum, pmax);
  k_pool_gather<<<g1((long)NG * 256), B, 0, stream>>>(psum, pmax, cnt, g16);

  // ---- regressor ----
  const int Gtiles = NG / 16;   // 256
  gemm_wmma<512><<<dim3(Gtiles / 4), B, 0, stream>>>(g16, regInT, rA, Gtiles);
  k_axpy_row<<<g1((long)NG * 256), B, 0, stream>>>(rA, Nvals, reg_in_W + (size_t)512 * 256);
  k_fill_f32<<<1, 256, 0, stream>>>(sums, 256, 0.f);
  k_fill_f32<<<1, 256, 0, stream>>>(sumsq, 256, 0.f);
  k_bn_stats<<<dim3((NG + 127) / 128), 256, 0, stream>>>(rA, NG, sums, sumsq);
  k_bn_final<<<1, 256, 0, stream>>>(sums, sumsq, (float)NG, reg_in_g, reg_in_beta, scale, shift);
  k_bn_apply<<<g1((long)NG * 256), B, 0, stream>>>(rA, scale, shift, (long)NG * 256, rA, r16);

  for (int i = 0; i < 3; ++i) {
    const float* hg = (const float*)d_in[37 + 4 * i + 2];
    const float* hb = (const float*)d_in[37 + 4 * i + 3];
    gemm_wmma<256><<<dim3(Gtiles / 4), B, 0, stream>>>(r16, regHT0 + (size_t)i * 65536, rA, Gtiles);
    k_fill_f32<<<1, 256, 0, stream>>>(sums, 256, 0.f);
    k_fill_f32<<<1, 256, 0, stream>>>(sumsq, 256, 0.f);
    k_bn_stats<<<dim3((NG + 127) / 128), 256, 0, stream>>>(rA, NG, sums, sumsq);
    k_bn_final<<<1, 256, 0, stream>>>(sums, sumsq, (float)NG, hg, hb, scale, shift);
    k_bn_apply<<<g1((long)NG * 256), B, 0, stream>>>(rA, scale, shift, (long)NG * 256, rA, r16);
  }

  k_reg_out<<<g1(NG), B, 0, stream>>>(rA, reg_out_W, reg_out_b, out);
}
